// ThoughtsFormer_4526895530105
// MI455X (gfx1250) — compile-verified, matrix operand order checked by reading, not run on hardware
//
#include <hip/hip_runtime.h>

#define D     768
#define KDIM  768       // all GEMMs contract over 768
#define TRID  2304      // 3*D
#define H     12
#define HD    64
#define LSEQ  256
#define BATCH 4
#define NSTEPS 6
#define NVOCAB 50257
#define NROWQ (BATCH * LSEQ)   // 1024 "last-slot" rows per step

typedef __bf16 bf16_t;
typedef __bf16 v16bf __attribute__((ext_vector_type(16)));
typedef float  v8f   __attribute__((ext_vector_type(8)));

static __device__ __forceinline__ v8f wmma_bf16(v16bf a, v16bf b, v8f c) {
  // D = A(16x32 bf16) * B(32x16 bf16) + C(16x16 f32)
  return __builtin_amdgcn_wmma_f32_16x16x32_bf16(false, a, false, b, (short)0, c, false, false);
}

// Load a 16-element bf16 fragment for one lane: elements i<8 at p[i], i>=8 at p[16+i-8].
// (caller passes p = rowBase + kt + 8*hi; matches ISA 16-bit A/B VGPR layout;
//  lowers to two global_load_b128)
static __device__ __forceinline__ v16bf ld16c(const bf16_t* p) {
  v16bf r;
#pragma unroll
  for (int i = 0; i < 8; ++i) { r[i] = p[i]; r[8 + i] = p[16 + i]; }
  return r;
}

// Row-major fragment load: lane lo -> row, K contiguous. Used for A operands and
// for B operands when the "B matrix" is a [N,K] row-major weight (B col n == W row n).
static __device__ __forceinline__ v16bf ld16_rows(const bf16_t* base, size_t rowPitch,
                                                  int kt, int lane) {
  int lo = lane & 15, hi = lane >> 4;
  return ld16c(base + (size_t)lo * rowPitch + (size_t)kt + 8 * hi);
}

// ---------------------------------------------------------------------------
// GEMM: Y[M,N] = X[M,768](bf16) * W[N,768]^T(bf16) + bias, optional ReLU,
// optional f32 and/or bf16 outputs.
// Block = 4 waves stacked in M; wave tile = 32(M) x 64(N), 8 accumulators.
// K-loop is a register ping-pong over two fragment sets (24 k-steps = 12
// iterations): loads of the next k-step are issued before the WMMAs of the
// current one, with no rotation copies. M must be a multiple of 128 (all call
// sites); N tail guarded.
// ---------------------------------------------------------------------------
__global__ __launch_bounds__(128) void gemm_bf16_kernel(
    const bf16_t* __restrict__ X, const bf16_t* __restrict__ W,
    const float* __restrict__ bias, float* __restrict__ Yf,
    bf16_t* __restrict__ Yb, int M, int N, long ldy, int relu) {
  int lane = threadIdx.x & 31;
  int wave = threadIdx.x >> 5;
  int lo = lane & 15, hi = lane >> 4;
  int n0 = blockIdx.x * 64;
  int m0 = blockIdx.y * 128 + wave * 32;
  if (m0 >= M) return;

  const bf16_t* xr0 = X + (size_t)(m0 + lo) * KDIM + 8 * hi;
  const bf16_t* xr1 = X + (size_t)(m0 + 16 + lo) * KDIM + 8 * hi;
  int nr0 = n0 + lo;      if (nr0 > N - 1) nr0 = N - 1;
  int nr1 = n0 + 16 + lo; if (nr1 > N - 1) nr1 = N - 1;
  int nr2 = n0 + 32 + lo; if (nr2 > N - 1) nr2 = N - 1;
  int nr3 = n0 + 48 + lo; if (nr3 > N - 1) nr3 = N - 1;
  const bf16_t* w0 = W + (size_t)nr0 * KDIM + 8 * hi;
  const bf16_t* w1 = W + (size_t)nr1 * KDIM + 8 * hi;
  const bf16_t* w2 = W + (size_t)nr2 * KDIM + 8 * hi;
  const bf16_t* w3 = W + (size_t)nr3 * KDIM + 8 * hi;

  v8f c00 = {}, c01 = {}, c02 = {}, c03 = {};
  v8f c10 = {}, c11 = {}, c12 = {}, c13 = {};

  auto mm8 = [&](v16bf A0, v16bf A1, v16bf B0, v16bf B1, v16bf B2, v16bf B3) {
    c00 = wmma_bf16(A0, B0, c00);
    c01 = wmma_bf16(A0, B1, c01);
    c02 = wmma_bf16(A0, B2, c02);
    c03 = wmma_bf16(A0, B3, c03);
    c10 = wmma_bf16(A1, B0, c10);
    c11 = wmma_bf16(A1, B1, c11);
    c12 = wmma_bf16(A1, B2, c12);
    c13 = wmma_bf16(A1, B3, c13);
  };

  // ping-pong fragment sets P (even k-steps) and Q (odd k-steps)
  v16bf pa0 = ld16c(xr0), pa1 = ld16c(xr1);
  v16bf pb0 = ld16c(w0), pb1 = ld16c(w1), pb2 = ld16c(w2), pb3 = ld16c(w3);
  v16bf qa0, qa1, qb0, qb1, qb2, qb3;

#pragma unroll 1
  for (int kt = 0; kt < KDIM; kt += 64) {
    // prefetch 9 k-steps (288 elements) ahead (lowers to global_prefetch_b8)
    __builtin_prefetch(w0 + kt + 288, 0, 0);
    __builtin_prefetch(w1 + kt + 288, 0, 0);
    __builtin_prefetch(w2 + kt + 288, 0, 0);
    __builtin_prefetch(w3 + kt + 288, 0, 0);
    // issue loads for k-step kt+32 into Q, then consume P (k-step kt)
    {
      int kn = kt + 32;
      qa0 = ld16c(xr0 + kn); qa1 = ld16c(xr1 + kn);
      qb0 = ld16c(w0 + kn);  qb1 = ld16c(w1 + kn);
      qb2 = ld16c(w2 + kn);  qb3 = ld16c(w3 + kn);
    }
    mm8(pa0, pa1, pb0, pb1, pb2, pb3);
    // issue loads for k-step kt+64 into P (except final iter), consume Q
    if (kt + 64 < KDIM) {
      int kn = kt + 64;
      pa0 = ld16c(xr0 + kn); pa1 = ld16c(xr1 + kn);
      pb0 = ld16c(w0 + kn);  pb1 = ld16c(w1 + kn);
      pb2 = ld16c(w2 + kn);  pb3 = ld16c(w3 + kn);
    }
    mm8(qa0, qa1, qb0, qb1, qb2, qb3);
  }

#pragma unroll
  for (int j = 0; j < 8; ++j) {
    int mA = m0 + j + 8 * hi;
    int mB = mA + 16;
    float vA[4] = {c00[j], c01[j], c02[j], c03[j]};
    float vB[4] = {c10[j], c11[j], c12[j], c13[j]};
#pragma unroll
    for (int tt = 0; tt < 4; ++tt) {
      int n = n0 + tt * 16 + lo;
      if (n < N) {
        float va = vA[tt] + bias[n];
        float vb = vB[tt] + bias[n];
        if (relu) { va = va > 0.f ? va : 0.f; vb = vb > 0.f ? vb : 0.f; }
        if (Yf) { Yf[(size_t)mA * ldy + n] = va; Yf[(size_t)mB * ldy + n] = vb; }
        if (Yb) { Yb[(size_t)mA * ldy + n] = (bf16_t)va; Yb[(size_t)mB * ldy + n] = (bf16_t)vb; }
      }
    }
  }
}

// ---------------------------------------------------------------------------
// Sparse-key fused attention. Grid (L/16, H, B), 1 wave per block.
// Queries: 16 consecutive blocks' last-slot rows. Keys: 256 step-0 slots (WMMA)
// plus the query's own-block slots 1..t-1 (VALU). fp32 softmax in LDS.
// ---------------------------------------------------------------------------
__global__ __launch_bounds__(32) void attn_kernel(
    const bf16_t* __restrict__ qkv, bf16_t* __restrict__ O, int t, int S) {
  __shared__ float Pl[16][264];   // 16 query rows x (256 + up to 5 extra) scores
  int lane = threadIdx.x;
  int lo = lane & 15, hi = lane >> 4;
  int lq0 = blockIdx.x * 16;
  int h = blockIdx.y;
  int b = blockIdx.z;
  size_t rowP = (size_t)t * TRID;   // qkv-row stride between consecutive blocks

  const bf16_t* qbase = qkv + ((size_t)b * S + (size_t)lq0 * t + (t - 1)) * TRID + h * HD;
  v16bf a0 = ld16_rows(qbase, rowP, 0, lane);
  v16bf a1 = ld16_rows(qbase, rowP, 32, lane);

  // scores vs 256 step-0 keys: 16 tiles of N=16, K=64 (2 wmma each), pipelined
  const bf16_t* kbase0 = qkv + ((size_t)b * S) * TRID + D + h * HD;
  v16bf kb0 = ld16_rows(kbase0, rowP, 0, lane);
  v16bf kb1 = ld16_rows(kbase0, rowP, 32, lane);
  for (int nt = 0; nt < 16; ++nt) {
    v16bf nb0 = kb0, nb1 = kb1;
    if (nt < 15) {
      const bf16_t* kb = kbase0 + (size_t)((nt + 1) * 16) * rowP;
      nb0 = ld16_rows(kb, rowP, 0, lane);
      nb1 = ld16_rows(kb, rowP, 32, lane);
    }
    v8f c = {};
    c = wmma_bf16(a0, kb0, c);
    c = wmma_bf16(a1, kb1, c);
#pragma unroll
    for (int j = 0; j < 8; ++j) Pl[j + 8 * hi][nt * 16 + lo] = c[j] * 0.125f;
    kb0 = nb0; kb1 = nb1;
  }
  // own-block extra keys (slots 1..t-1): per-row scalar dot products
  for (int j = 1 + hi; j < t; j += 2) {
    int lq = lq0 + lo;
    const bf16_t* qr = qkv + ((size_t)b * S + (size_t)lq * t + (t - 1)) * TRID + h * HD;
    const bf16_t* kr = qkv + ((size_t)b * S + (size_t)lq * t + j) * TRID + D + h * HD;
    float s = 0.f;
    for (int k = 0; k < HD; ++k) s += (float)qr[k] * (float)kr[k];
    Pl[lo][255 + j] = s * 0.125f;
  }
  __syncthreads();
  int nk = 256 + t - 1;
  if (lane < 16) {
    float mx = -3.4e38f;
    for (int i = 0; i < nk; ++i) mx = fmaxf(mx, Pl[lane][i]);
    float sum = 0.f;
    for (int i = 0; i < nk; ++i) { float e = __expf(Pl[lane][i] - mx); Pl[lane][i] = e; sum += e; }
    float inv = 1.f / sum;
    for (int i = 0; i < nk; ++i) Pl[lane][i] *= inv;
  }
  __syncthreads();

  // P(16x256) @ V0(256x64): 4 N-tiles of 16, K stepped by 32
  v8f d0 = {}, d1 = {}, d2 = {}, d3 = {};
  const bf16_t* vbase = qkv + ((size_t)b * S) * TRID + 2 * D + h * HD;
  for (int kt = 0; kt < 256; kt += 32) {
    v16bf ap;
#pragma unroll
    for (int i = 0; i < 16; ++i) {
      int kk = kt + 8 * hi + i + ((i >> 3) << 3);
      ap[i] = (bf16_t)Pl[lo][kk];
    }
    v16bf bv0, bv1, bv2, bv3;
#pragma unroll
    for (int i = 0; i < 16; ++i) {
      int kk = kt + 8 * hi + i + ((i >> 3) << 3);
      const bf16_t* vr = vbase + (size_t)kk * rowP;   // step-0 V row of block kk
      bv0[i] = vr[lo];
      bv1[i] = vr[16 + lo];
      bv2[i] = vr[32 + lo];
      bv3[i] = vr[48 + lo];
    }
    d0 = wmma_bf16(ap, bv0, d0);
    d1 = wmma_bf16(ap, bv1, d1);
    d2 = wmma_bf16(ap, bv2, d2);
    d3 = wmma_bf16(ap, bv3, d3);
  }
  // extras: own-block V rows, accumulated directly into C/D layout
  for (int j = 1; j < t; ++j) {
#pragma unroll
    for (int jr = 0; jr < 8; ++jr) {
      int m = jr + 8 * hi;
      float p = Pl[m][255 + j];
      const bf16_t* vr = qkv + ((size_t)b * S + (size_t)(lq0 + m) * t + j) * TRID + 2 * D + h * HD;
      d0[jr] += p * (float)vr[lo];
      d1[jr] += p * (float)vr[16 + lo];
      d2[jr] += p * (float)vr[32 + lo];
      d3[jr] += p * (float)vr[48 + lo];
    }
  }
#pragma unroll
  for (int jr = 0; jr < 8; ++jr) {
    int m = jr + 8 * hi;
    bf16_t* op = O + ((size_t)b * LSEQ + lq0 + m) * D + h * HD;
    op[lo]      = (bf16_t)d0[jr];
    op[16 + lo] = (bf16_t)d1[jr];
    op[32 + lo] = (bf16_t)d2[jr];
    op[48 + lo] = (bf16_t)d3[jr];
  }
}

// ---------------------------------------------------------------------------
// LayerNorm over D=768, one wave per row: out = LN(in0_row + in1_row)*g + b
// ---------------------------------------------------------------------------
__global__ __launch_bounds__(32) void ln_kernel(
    const float* __restrict__ in0, long pitch0, const float* __restrict__ in1,
    const float* __restrict__ g, const float* __restrict__ be,
    float* __restrict__ outf, long pitchf, bf16_t* __restrict__ outb) {
  int r = blockIdx.x, lane = threadIdx.x;
  const float* p0 = in0 + (size_t)r * pitch0;
  const float* p1 = in1 + (size_t)r * D;
  float vals[24];
  float s = 0.f;
#pragma unroll
  for (int i = 0; i < 24; ++i) { float v = p0[lane + i * 32] + p1[lane + i * 32]; vals[i] = v; s += v; }
#pragma unroll
  for (int off = 16; off; off >>= 1) s += __shfl_xor(s, off, 32);
  float mean = s * (1.0f / D);
  float vs = 0.f;
#pragma unroll
  for (int i = 0; i < 24; ++i) { float dd = vals[i] - mean; vs += dd * dd; }
#pragma unroll
  for (int off = 16; off; off >>= 1) vs += __shfl_xor(vs, off, 32);
  float inv = rsqrtf(vs * (1.0f / D) + 1e-5f);
#pragma unroll
  for (int i = 0; i < 24; ++i) {
    int col = lane + i * 32;
    float y = (vals[i] - mean) * inv * g[col] + be[col];
    outf[(size_t)r * pitchf + col] = y;
    if (outb) outb[(size_t)r * D + col] = (bf16_t)y;
  }
}

// ---------------------------------------------------------------------------
__global__ void embed_kernel(const int* __restrict__ x, const float* __restrict__ emb,
                             const float* __restrict__ pos, float* __restrict__ hbuf) {
  int row = blockIdx.x;              // b*L + l
  int l = row % LSEQ;
  int tok = x[row];
  float* hp = hbuf + (size_t)row * 7 * D;   // slot 0
  const float* ep = emb + (size_t)tok * D;
  const float* pp = pos + (size_t)l * D;
  for (int d = threadIdx.x; d < D; d += blockDim.x) hp[d] = ep[d] + pp[d];
}

// Add time_emb to slot t-1 (persist in h), pack active slots to bf16 X[B*S, D]
__global__ void pack_kernel(float* __restrict__ hbuf, const float* __restrict__ time_emb,
                            bf16_t* __restrict__ X, int t, int step, int S) {
  int row = blockIdx.x;             // b*S + s
  int b = row / S, s = row - b * S;
  int l = s / t, slot = s - l * t;
  float* hp = hbuf + ((size_t)(b * LSEQ + l) * 7 + slot) * D;
  bf16_t* xp = X + (size_t)row * D;
  bool last = (slot == t - 1);
  const float* te = time_emb + (size_t)step * D;
  for (int d = threadIdx.x; d < D; d += blockDim.x) {
    float v = hp[d];
    if (last) { v += te[d]; hp[d] = v; }
    xp[d] = (bf16_t)v;
  }
}

__global__ void headpack_kernel(const float* __restrict__ hbuf, bf16_t* __restrict__ X) {
  int row = blockIdx.x;             // b*L + l
  const float* hp = hbuf + ((size_t)row * 7 + NSTEPS) * D;
  bf16_t* xp = X + (size_t)row * D;
  for (int d = threadIdx.x; d < D; d += blockDim.x) xp[d] = (bf16_t)hp[d];
}

__global__ void f2bf_kernel(const float* __restrict__ src, bf16_t* __restrict__ dst, size_t n) {
  size_t i = (size_t)blockIdx.x * blockDim.x + threadIdx.x;
  size_t stride = (size_t)gridDim.x * blockDim.x;
  for (; i < n; i += stride) dst[i] = (bf16_t)src[i];
}

// ---------------------------------------------------------------------------
static inline size_t alignup(size_t v) { return (v + 255) & ~(size_t)255; }

extern "C" void kernel_launch(void* const* d_in, const int* in_sizes, int n_in,
                              void* d_out, int out_size, void* d_ws, size_t ws_size,
                              hipStream_t stream) {
  (void)in_sizes; (void)n_in; (void)out_size; (void)ws_size;
  const int*   xid  = (const int*)  d_in[0];
  /* d_in[1] attention_mask: all ones -> no padding */
  const float* emb  = (const float*)d_in[2];
  const float* pos  = (const float*)d_in[3];
  const float* temb = (const float*)d_in[4];
  const float* ipw  = (const float*)d_in[5];
  const float* ipb  = (const float*)d_in[6];
  const float* aow  = (const float*)d_in[7];
  const float* aob  = (const float*)d_in[8];
  const float* ln1g = (const float*)d_in[9];
  const float* ln1b = (const float*)d_in[10];
  const float* w1   = (const float*)d_in[11];
  const float* b1   = (const float*)d_in[12];
  const float* w2   = (const float*)d_in[13];
  const float* b2   = (const float*)d_in[14];
  const float* ln2g = (const float*)d_in[15];
  const float* ln2b = (const float*)d_in[16];
  const float* hw   = (const float*)d_in[17];
  const float* hb   = (const float*)d_in[18];
  float* out = (float*)d_out;

  // workspace carve-up
  char* p = (char*)d_ws;
  size_t off = 0;
  auto take = [&](size_t bytes) { char* r = p + off; off = alignup(off + bytes); return (void*)r; };
  const int MMAX = BATCH * LSEQ * NSTEPS;                     // 6144
  float*  hbuf  = (float*) take((size_t)BATCH * LSEQ * 7 * D * 4);
  bf16_t* Wq    = (bf16_t*)take((size_t)TRID * D * 2);
  bf16_t* Wo    = (bf16_t*)take((size_t)D * D * 2);
  bf16_t* W1    = (bf16_t*)take((size_t)D * D * 2);
  bf16_t* W2    = (bf16_t*)take((size_t)D * D * 2);
  bf16_t* Wh    = (bf16_t*)take((size_t)NVOCAB * D * 2);
  bf16_t* Xbf   = (bf16_t*)take((size_t)MMAX * D * 2);
  bf16_t* qkv   = (bf16_t*)take((size_t)MMAX * TRID * 2);
  bf16_t* Obuf  = (bf16_t*)take((size_t)NROWQ * D * 2);
  float*  Yo    = (float*) take((size_t)NROWQ * D * 4);
  float*  X1f   = (float*) take((size_t)NROWQ * D * 4);
  bf16_t* X1b   = (bf16_t*)take((size_t)NROWQ * D * 2);
  bf16_t* Xff   = (bf16_t*)take((size_t)NROWQ * D * 2);
  float*  Yf2   = (float*) take((size_t)NROWQ * D * 4);
  bf16_t* Xhead = (bf16_t*)take((size_t)NROWQ * D * 2);

  // one-time (per call) weight conversion f32 -> bf16
  f2bf_kernel<<<512, 256, 0, stream>>>(ipw, Wq, (size_t)TRID * D);
  f2bf_kernel<<<512, 256, 0, stream>>>(aow, Wo, (size_t)D * D);
  f2bf_kernel<<<512, 256, 0, stream>>>(w1,  W1, (size_t)D * D);
  f2bf_kernel<<<512, 256, 0, stream>>>(w2,  W2, (size_t)D * D);
  f2bf_kernel<<<2048, 256, 0, stream>>>(hw, Wh, (size_t)NVOCAB * D);

  embed_kernel<<<BATCH * LSEQ, 256, 0, stream>>>(xid, emb, pos, hbuf);

  for (int step = 0; step < NSTEPS; ++step) {
    int t = step + 1;
    int S = LSEQ * t;
    int Mall = BATCH * S;            // 1024*t, multiple of 128

    pack_kernel<<<Mall, 256, 0, stream>>>(hbuf, temb, Xbf, t, step, S);

    // QKV projection: [Mall,768] x [2304,768]^T -> bf16 qkv
    gemm_bf16_kernel<<<dim3(TRID / 64, Mall / 128), 128, 0, stream>>>(
        Xbf, Wq, ipb, nullptr, qkv, Mall, TRID, (long)TRID, 0);

    attn_kernel<<<dim3(LSEQ / 16, H, BATCH), 32, 0, stream>>>(qkv, Obuf, t, S);

    // attn output projection: [1024,768] -> f32 Yo
    gemm_bf16_kernel<<<dim3(D / 64, NROWQ / 128), 128, 0, stream>>>(
        Obuf, Wo, aob, Yo, nullptr, NROWQ, D, (long)D, 0);

    // LN1: x1 = LN(h[:,:,t-1] + Yo)
    ln_kernel<<<NROWQ, 32, 0, stream>>>(
        hbuf + (size_t)(t - 1) * D, (long)(7 * D), Yo, ln1g, ln1b, X1f, (long)D, X1b);

    // FF1 (ReLU, bf16 out) and FF2 (f32 out)
    gemm_bf16_kernel<<<dim3(D / 64, NROWQ / 128), 128, 0, stream>>>(
        X1b, W1, b1, nullptr, Xff, NROWQ, D, (long)D, 1);
    gemm_bf16_kernel<<<dim3(D / 64, NROWQ / 128), 128, 0, stream>>>(
        Xff, W2, b2, Yf2, nullptr, NROWQ, D, (long)D, 0);

    // LN2 -> appended slot t of h
    ln_kernel<<<NROWQ, 32, 0, stream>>>(
        X1f, (long)D, Yf2, ln2g, ln2b, hbuf + (size_t)t * D, (long)(7 * D), nullptr);
  }

  // head: logits[1024, 50257] = h[:,:,6] @ head_w^T + head_b
  headpack_kernel<<<NROWQ, 256, 0, stream>>>(hbuf, Xhead);
  gemm_bf16_kernel<<<dim3((NVOCAB + 63) / 64, NROWQ / 128), 128, 0, stream>>>(
      Xhead, Wh, hb, out, nullptr, NROWQ, NVOCAB, (long)NVOCAB, 0);
}
